// UPPT_Decoder_84215718740286
// MI455X (gfx1250) — compile-verified, matrix operand order checked by reading, not run on hardware
//
#include <hip/hip_runtime.h>
#include <stdint.h>

typedef __bf16 bf16_t;
typedef __attribute__((ext_vector_type(16))) __bf16 v16bf;
typedef __attribute__((ext_vector_type(8)))  float  v8f;
typedef __attribute__((ext_vector_type(4)))  unsigned int v4u;

#define cB 64
#define cT 500
#define cE 256
#define cG 256
#define cA 128
#define cD 400
#define cH1 256
#define cH2 128
#define cS 500
#define KP1 416     // K=400 padded to multiple of 32
#define ROWS 8      // batch rows per workgroup (8 WGs x 8 rows)

// ---------------- WMMA helpers ----------------

__device__ __forceinline__ v8f wmma_bf16(v16bf a, v16bf b, v8f c) {
  return __builtin_amdgcn_wmma_f32_16x16x32_bf16(false, a, false, b, (short)0, c, false, false);
}

// A-fragment from LDS bf16 row-major [16][Kpad]; ISA 16-bit A layout:
// lane m=lane&15; VGPR0-3: K = (lane<16?0:8)+0..7 ; VGPR4-7: K = 16+(lane<16?0:8)+0..7
__device__ __forceinline__ v16bf load_A_lds(const bf16_t* sa, int Kpad, int kb, int lane) {
  const int m  = lane & 15;
  const int hi = (lane >> 4) & 1;
  const bf16_t* p = sa + m * Kpad + kb * 32 + hi * 8;
  union { v16bf v; v4u u[2]; } r;
  r.u[0] = *(const v4u*)(p);
  r.u[1] = *(const v4u*)(p + 16);
  return r.v;
}

// B-fragment from pre-swizzled global weights: [kblk][ntile][lane][16] contiguous
__device__ __forceinline__ v16bf load_B_glb(const bf16_t* w, int nn, int kb, int nt, int lane) {
  const bf16_t* p = w + ((((size_t)kb * nn + nt) * 32 + lane) << 4);
  union { v16bf v; v4u u[2]; } r;
  const v4u* q = (const v4u*)p;
  r.u[0] = q[0];
  r.u[1] = q[1];
  return r.v;
}

__device__ __forceinline__ float fsigmoid(float x) { return 1.f / (1.f + __expf(-x)); }
__device__ __forceinline__ float ftanh(float x) {
  x = fminf(15.f, fmaxf(-15.f, x));
  const float e = __expf(2.f * x);
  return (e - 1.f) / (e + 1.f);
}

// gfx1250 hardware tanh (TRANS op). Four back-to-back tanhs: each serves as the
// required 1-independent-op spacer for the previous; trailing v_nop covers the last.
__device__ __forceinline__ void ftanh4(float& x, float& y, float& z, float& w) {
  asm volatile("v_tanh_f32 %0, %0\n\t"
               "v_tanh_f32 %1, %1\n\t"
               "v_tanh_f32 %2, %2\n\t"
               "v_tanh_f32 %3, %3\n\t"
               "v_nop"
               : "+v"(x), "+v"(y), "+v"(z), "+v"(w));
}

// ---------------- weight swizzle: f32 [K,N] -> bf16 WMMA B fragments ----------------

__global__ void swizzle_w(const float* __restrict__ W, bf16_t* __restrict__ dst,
                          int Korig, int Kpad, int N, int total) {
  const int idx = blockIdx.x * 256 + threadIdx.x;
  if (idx >= total) return;
  const int e    = idx & 15;
  const int lane = (idx >> 4) & 31;
  const int t    = idx >> 9;
  const int nn   = N >> 4;
  const int nt   = t % nn;
  const int ik   = t / nn;
  const int j    = e >> 1, half = e & 1;
  const int hi   = (lane >> 4) & 1;
  const int kofs = (j < 4) ? (hi * 8 + 2 * j + half) : (16 + hi * 8 + 2 * (j - 4) + half);
  const int k    = ik * 32 + kofs;
  const int n    = nt * 16 + (lane & 15);
  const float v  = (k < Korig) ? W[(size_t)k * N + n] : 0.f;
  dst[idx] = (bf16_t)v;
}

// ---------------- enc_proj = enc_feat[B*T,E] @ W_enc[E,A]  (WMMA GEMM) ----------------

__global__ __launch_bounds__(256) void enc_proj_kernel(
    const float* __restrict__ enc_feat, const bf16_t* __restrict__ Wencs,
    float* __restrict__ enc_proj) {
  const int lane = threadIdx.x & 31;
  const int wv   = threadIdx.x >> 5;   // N tile 0..7
  const int mt   = blockIdx.x;         // M tile 0..1999
  const int m    = lane & 15;
  const int hi   = (lane >> 4) & 1;
  const size_t row = (size_t)mt * 16 + m;
  v8f c = {};
#pragma unroll
  for (int kb = 0; kb < 8; ++kb) {
    const float* p = enc_feat + row * cE + kb * 32 + hi * 8;
    v16bf a;
#pragma unroll
    for (int j = 0; j < 8; ++j) a[j] = (bf16_t)p[j];
#pragma unroll
    for (int j = 0; j < 8; ++j) a[8 + j] = (bf16_t)p[16 + j];
    c = wmma_bf16(a, load_B_glb(Wencs, 8, kb, wv, lane), c);
  }
  const int n = wv * 16 + (lane & 15);
  const int rbase = (lane < 16) ? 0 : 8;
#pragma unroll
  for (int r = 0; r < 8; ++r)
    enc_proj[((size_t)mt * 16 + rbase + r) * cA + n] = c[r];
}

// ---------------- persistent decoder: 8 WGs x 8 batch rows, 500 steps ----------------
// WMMA tiles remain M=16; rows ROWS..15 hold don't-care data that only ever
// propagates to output rows >= ROWS (A-row m only affects D-row m) and is never stored.

constexpr size_t SM_BF16  = (size_t)16 * (KP1 + 256 + 384 + 256 + 512);             // elems
constexpr size_t SM_F32   = (size_t)16 * 768 + 16 * 256 + 16 * 256 + 16 * 128
                          + (size_t)16 * 512 + 16 * 400 + 128;                       // elems
constexpr size_t SMEM_BYTES = SM_BF16 * 2 + SM_F32 * 4;                              // 207360

__global__ __launch_bounds__(512, 1) void decoder_persistent(
    const float* __restrict__ enc_feat,
    const float* __restrict__ b_p1, const float* __restrict__ b_p2,
    const float* __restrict__ b_ix, const float* __restrict__ b_ih,
    const float* __restrict__ b_attn, const float* __restrict__ v_attn,
    const float* __restrict__ b_out,
    const bf16_t* __restrict__ Wp1s, const bf16_t* __restrict__ Wp2s,
    const bf16_t* __restrict__ Wixs, const bf16_t* __restrict__ Wihs,
    const bf16_t* __restrict__ Wdecs, const bf16_t* __restrict__ Wouts,
    const float* __restrict__ enc_proj,
    float* __restrict__ out_pred, float* __restrict__ out_attn) {
  extern __shared__ char smem[];
  bf16_t* sh_pred = (bf16_t*)smem;               // [16][416] bf16
  bf16_t* sh_x1   = sh_pred + 16 * KP1;          // [16][256]
  bf16_t* sh_gin  = sh_x1 + 16 * 256;            // [16][384] = [x2 | ctx]
  bf16_t* sh_hbf  = sh_gin + 16 * 384;           // [16][256]
  bf16_t* sh_hc   = sh_hbf + 16 * 256;           // [16][512] = [h | ctx]
  float*  sh_g    = (float*)(sh_hc + 16 * 512);  // [16][768] gates (gi + gh r,z)
  float*  sh_ghn  = sh_g + 16 * 768;             // [16][256] gh n-part
  float*  sh_h    = sh_ghn + 16 * 256;           // [16][256] f32 h
  float*  sh_q    = sh_h + 16 * 256;             // [16][128] attn query
  float*  sh_at   = sh_q + 16 * 128;             // [16][512] scores/attn
  float*  sh_raw  = sh_at + 16 * 512;            // [16][400] logits
  float*  sh_v    = sh_raw + 16 * 400;           // [128] v_attn

  const int tid   = threadIdx.x;
  const int lane  = tid & 31;
  const int wv    = tid >> 5;             // 16 waves
  const int ln15  = lane & 15;
  const int rbase = (lane < 16) ? 0 : 8;
  const int b0    = blockIdx.x * ROWS;

  for (int i = tid; i < (int)(SMEM_BYTES / 4); i += 512) ((unsigned int*)smem)[i] = 0u;
  __syncthreads();
  if (tid < cA) sh_v[tid] = v_attn[tid];
  __syncthreads();

  for (int s = 0; s < cS; ++s) {
    // P1: x1 = relu(pred @ W_p1 + b_p1), N=256, K=416
    {
      const int nt = wv;
      v8f c = {};
#pragma unroll
      for (int kb = 0; kb < 13; ++kb)
        c = wmma_bf16(load_A_lds(sh_pred, KP1, kb, lane), load_B_glb(Wp1s, 16, kb, nt, lane), c);
      const int n = nt * 16 + ln15;
      const float bias = b_p1[n];
#pragma unroll
      for (int r = 0; r < 8; ++r) {
        const float v = c[r] + bias;
        sh_x1[(rbase + r) * 256 + n] = (bf16_t)(v > 0.f ? v : 0.f);
      }
    }
    __syncthreads();
    // P2: x2 = relu(x1 @ W_p2 + b_p2), N=128 -> sh_gin[:,0:128]
    if (wv < 8) {
      const int nt = wv;
      v8f c = {};
#pragma unroll
      for (int kb = 0; kb < 8; ++kb)
        c = wmma_bf16(load_A_lds(sh_x1, 256, kb, lane), load_B_glb(Wp2s, 8, kb, nt, lane), c);
      const int n = nt * 16 + ln15;
      const float bias = b_p2[n];
#pragma unroll
      for (int r = 0; r < 8; ++r) {
        const float v = c[r] + bias;
        sh_gin[(rbase + r) * 384 + n] = (bf16_t)(v > 0.f ? v : 0.f);
      }
    }
    __syncthreads();
    // P3a: gi = [x2,ctx] @ W_ix + b_ix, N=768, K=384
#pragma unroll
    for (int p = 0; p < 3; ++p) {
      const int nt = wv + 16 * p;
      v8f c = {};
#pragma unroll
      for (int kb = 0; kb < 12; ++kb)
        c = wmma_bf16(load_A_lds(sh_gin, 384, kb, lane), load_B_glb(Wixs, 48, kb, nt, lane), c);
      const int n = nt * 16 + ln15;
      const float bias = b_ix[n];
#pragma unroll
      for (int r = 0; r < 8; ++r)
        sh_g[(rbase + r) * 768 + n] = c[r] + bias;
    }
    __syncthreads();
    // P3b: gh = h @ W_ih + b_ih (r,z summed into sh_g; n-part kept separate)
#pragma unroll
    for (int p = 0; p < 3; ++p) {
      const int nt = wv + 16 * p;
      v8f c = {};
#pragma unroll
      for (int kb = 0; kb < 8; ++kb)
        c = wmma_bf16(load_A_lds(sh_hbf, 256, kb, lane), load_B_glb(Wihs, 48, kb, nt, lane), c);
      const int n = nt * 16 + ln15;
      const float bias = b_ih[n];
      if (n < 512) {
#pragma unroll
        for (int r = 0; r < 8; ++r) sh_g[(rbase + r) * 768 + n] += c[r] + bias;
      } else {
#pragma unroll
        for (int r = 0; r < 8; ++r) sh_ghn[(rbase + r) * 256 + (n - 512)] = c[r] + bias;
      }
    }
    __syncthreads();
    // P3c: GRU gates elementwise (only real rows matter; do all 16 for simplicity)
    for (int idx = tid; idx < 16 * 256; idx += 512) {
      const int m = idx >> 8, j = idx & 255;
      const float gr = sh_g[m * 768 + j];
      const float gz = sh_g[m * 768 + 256 + j];
      const float gn = sh_g[m * 768 + 512 + j];
      const float hn = sh_ghn[m * 256 + j];
      const float rg = fsigmoid(gr);
      const float zg = fsigmoid(gz);
      const float ng = ftanh(gn + rg * hn);
      const float h  = (1.f - zg) * ng + zg * sh_h[m * 256 + j];
      sh_h[m * 256 + j] = h;
      const bf16_t hb = (bf16_t)h;
      sh_hbf[m * 256 + j] = hb;
      sh_hc[m * 512 + j]  = hb;
    }
    __syncthreads();
    // P4: q = h @ W_dec + b_attn, N=128
    if (wv < 8) {
      const int nt = wv;
      v8f c = {};
#pragma unroll
      for (int kb = 0; kb < 8; ++kb)
        c = wmma_bf16(load_A_lds(sh_hbf, 256, kb, lane), load_B_glb(Wdecs, 8, kb, nt, lane), c);
      const int n = nt * 16 + ln15;
      const float bias = b_attn[n];
#pragma unroll
      for (int r = 0; r < 8; ++r) sh_q[(rbase + r) * 128 + n] = c[r] + bias;
    }
    __syncthreads();
    // P5: scores[m,t] = sum_a v[a]*tanh(enc_proj + q)
    // 2 waves per row: wave w -> row w>>1, t-range offset (w&1)*32, stride 64.
    {
      const int m  = wv >> 1;
      const int th = wv & 1;
      const float4* qp = (const float4*)(sh_q + m * 128);
      const float4* vp = (const float4*)sh_v;
      const float* epb = enc_proj + (size_t)(b0 + m) * cT * cA;
      for (int t = th * 32 + lane; t < cT; t += 64) {
        const float4* ep = (const float4*)(epb + (size_t)t * cA);
        float acc = 0.f;
#pragma unroll 4
        for (int i = 0; i < 32; ++i) {
          const float4 e4 = ep[i], q4 = qp[i], v4 = vp[i];
          float ux = e4.x + q4.x, uy = e4.y + q4.y, uz = e4.z + q4.z, uw = e4.w + q4.w;
          ftanh4(ux, uy, uz, uw);                  // v_tanh_f32 x4 (gfx1250 TRANS)
          acc += v4.x * ux + v4.y * uy + v4.z * uz + v4.w * uw;
        }
        sh_at[m * 512 + t] = acc;
      }
    }
    __syncthreads();
    // P5b: softmax over T per row (1 wave per row, 8 waves) + stream attn to output
    if (wv < ROWS) {
      const int m = wv;
      float mx = -1e30f;
      for (int t = lane; t < cT; t += 32) mx = fmaxf(mx, sh_at[m * 512 + t]);
#pragma unroll
      for (int o = 16; o > 0; o >>= 1) mx = fmaxf(mx, __shfl_xor(mx, o, 32));
      float sum = 0.f;
      for (int t = lane; t < cT; t += 32) {
        const float e = __expf(sh_at[m * 512 + t] - mx);
        sh_at[m * 512 + t] = e;
        sum += e;
      }
#pragma unroll
      for (int o = 16; o > 0; o >>= 1) sum += __shfl_xor(sum, o, 32);
      const float inv = 1.f / sum;
      float* oat = out_attn + ((size_t)(b0 + m) * cS + s) * cT;
      for (int t = lane; t < cT; t += 32) {
        const float pr = sh_at[m * 512 + t] * inv;
        sh_at[m * 512 + t] = pr;
        oat[t] = pr;
      }
    }
    __syncthreads();
    // P6: context[m,e] = sum_t attn[m,t]*enc_feat[b,t,e]
    // 2 waves per row: wave w -> row w>>1, E-half (w&1)*128; lane covers 4 e's.
    {
      const int m  = wv >> 1;
      const int e0 = (wv & 1) * 128 + lane * 4;
      const float* ef = enc_feat + (size_t)(b0 + m) * cT * cE + e0;
      float a0 = 0.f, a1 = 0.f, a2 = 0.f, a3 = 0.f;
      for (int t = 0; t < cT; ++t) {
        const float w = sh_at[m * 512 + t];
        const float4 x = *(const float4*)(ef + (size_t)t * cE);
        a0 += w * x.x; a1 += w * x.y; a2 += w * x.z; a3 += w * x.w;
      }
      bf16_t c0 = (bf16_t)a0, c1 = (bf16_t)a1, c2 = (bf16_t)a2, c3 = (bf16_t)a3;
      sh_gin[m * 384 + 128 + e0 + 0] = c0;  // next step's GRU input
      sh_gin[m * 384 + 128 + e0 + 1] = c1;
      sh_gin[m * 384 + 128 + e0 + 2] = c2;
      sh_gin[m * 384 + 128 + e0 + 3] = c3;
      sh_hc[m * 512 + 256 + e0 + 0]  = c0;  // this step's output proj
      sh_hc[m * 512 + 256 + e0 + 1]  = c1;
      sh_hc[m * 512 + 256 + e0 + 2]  = c2;
      sh_hc[m * 512 + 256 + e0 + 3]  = c3;
    }
    __syncthreads();
    // P7: raw = [h,ctx] @ W_out + b_out, N=400 (25 tiles), K=512
#pragma unroll
    for (int p = 0; p < 2; ++p) {
      const int nt = wv + 16 * p;
      if (nt < 25) {
        v8f c = {};
#pragma unroll
        for (int kb = 0; kb < 16; ++kb)
          c = wmma_bf16(load_A_lds(sh_hc, 512, kb, lane), load_B_glb(Wouts, 25, kb, nt, lane), c);
        const int n = nt * 16 + ln15;
        const float bias = b_out[n];
#pragma unroll
        for (int r = 0; r < 8; ++r) sh_raw[(rbase + r) * 400 + n] = c[r] + bias;
      }
    }
    __syncthreads();
    // P8: chunked softmax (5 chunks of 80) -> pred (global + bf16 LDS for next step)
    if (tid < ROWS * 5) {
      const int m = tid / 5, ch = tid % 5;
      float* rw = sh_raw + m * 400 + ch * 80;
      float mx = -1e30f;
      for (int j = 0; j < 80; ++j) mx = fmaxf(mx, rw[j]);
      float sum = 0.f;
      for (int j = 0; j < 80; ++j) { const float e = __expf(rw[j] - mx); rw[j] = e; sum += e; }
      const float inv = 1.f / sum;
      float* op = out_pred + ((size_t)(b0 + m) * cS + s) * cD + ch * 80;
      for (int j = 0; j < 80; ++j) {
        const float pr = rw[j] * inv;
        op[j] = pr;
        sh_pred[m * KP1 + ch * 80 + j] = (bf16_t)pr;
      }
    }
    __syncthreads();
  }
}

// ---------------- host side ----------------

extern "C" void kernel_launch(void* const* d_in, const int* in_sizes, int n_in,
                              void* d_out, int out_size, void* d_ws, size_t ws_size,
                              hipStream_t stream) {
  (void)in_sizes; (void)n_in; (void)out_size; (void)ws_size;
  const float* enc_feat = (const float*)d_in[0];
  const float* W_p1 = (const float*)d_in[1];
  const float* b_p1 = (const float*)d_in[2];
  const float* W_p2 = (const float*)d_in[3];
  const float* b_p2 = (const float*)d_in[4];
  const float* W_ix = (const float*)d_in[5];
  const float* W_ih = (const float*)d_in[6];
  const float* b_ix = (const float*)d_in[7];
  const float* b_ih = (const float*)d_in[8];
  const float* W_dec = (const float*)d_in[9];
  const float* W_enc = (const float*)d_in[10];
  const float* b_attn = (const float*)d_in[11];
  const float* v_attn = (const float*)d_in[12];
  const float* W_out = (const float*)d_in[13];
  const float* b_out = (const float*)d_in[14];

  char* ws = (char*)d_ws;
  size_t o = 0;
  bf16_t* Wp1s  = (bf16_t*)(ws + o); o += (size_t)KP1 * 256 * 2;
  bf16_t* Wp2s  = (bf16_t*)(ws + o); o += (size_t)256 * 128 * 2;
  bf16_t* Wixs  = (bf16_t*)(ws + o); o += (size_t)384 * 768 * 2;
  bf16_t* Wihs  = (bf16_t*)(ws + o); o += (size_t)256 * 768 * 2;
  bf16_t* Wdecs = (bf16_t*)(ws + o); o += (size_t)256 * 128 * 2;
  bf16_t* Wencs = (bf16_t*)(ws + o); o += (size_t)256 * 128 * 2;
  bf16_t* Wouts = (bf16_t*)(ws + o); o += (size_t)512 * 400 * 2;
  float*  enc_proj = (float*)(ws + o); o += (size_t)cB * cT * cA * 4;

  auto launch_sw = [&](const float* W, bf16_t* dst, int Ko, int Kp, int N) {
    const int total = Kp * N;
    swizzle_w<<<(total + 255) / 256, 256, 0, stream>>>(W, dst, Ko, Kp, N, total);
  };
  launch_sw(W_p1, Wp1s, 400, KP1, 256);
  launch_sw(W_p2, Wp2s, 256, 256, 128);
  launch_sw(W_ix, Wixs, 384, 384, 768);
  launch_sw(W_ih, Wihs, 256, 256, 768);
  launch_sw(W_dec, Wdecs, 256, 256, 128);
  launch_sw(W_enc, Wencs, 256, 256, 128);
  launch_sw(W_out, Wouts, 512, 512, 400);

  enc_proj_kernel<<<(cB * cT) / 16, 256, 0, stream>>>(enc_feat, Wencs, enc_proj);

  float* out_pred = (float*)d_out;
  float* out_attn = out_pred + (size_t)cB * cS * cD;
  decoder_persistent<<<cB / ROWS, 512, SMEM_BYTES, stream>>>(
      enc_feat, b_p1, b_p2, b_ix, b_ih, b_attn, v_attn, b_out,
      Wp1s, Wp2s, Wixs, Wihs, Wdecs, Wouts, enc_proj, out_pred, out_attn);
}
